// PCENetwork_58128087384140
// MI455X (gfx1250) — matrix operand order repaired and checked by reading, not compiled, not run on hardware
//
#include <hip/hip_runtime.h>
#include <hip/hip_bf16.h>
#include <math.h>

// ---------------- problem constants ----------------
#define BB        32
#define C_IN      3
#define HH        256
#define WW        256
#define PATCH     16
#define PP        256            // patches per image (16x16)
#define BP        (BB * PP)      // 8192 total patches
#define EE        8
#define C_OUT     8
#define NCH       (EE * C_OUT)   // 64 fused expert output channels
#define KK        27             // 3*3*3 im2col depth
#define C_PROJ    16
#define NUM_CLASSES 1000
#define SPP_DIM   168            // 8 * (1 + 4 + 16)
#define GATE_STEEP 10.0f

// haloed f16 patch: [3][18][18] + zero tail so K>=27 reads a guaranteed zero
#define HALO_CH   324            // 18*18
#define HALO_SZ   972            // 3*324
#define ZERO_SLOT 972
#define RAWH_SZ   1258           // 972 + 286 (max base 285)

typedef __attribute__((ext_vector_type(16))) _Float16 v16h;
typedef __attribute__((ext_vector_type(8)))  float    v8f;
typedef __attribute__((ext_vector_type(2)))  int      v2i;

// typed, addrspace-qualified pointers for the async-LDS builtin
typedef v2i __attribute__((address_space(1))) as1_v2i;
typedef v2i __attribute__((address_space(3))) as3_v2i;

#if defined(__has_builtin)
#  if __has_builtin(__builtin_amdgcn_global_load_async_to_lds_b64)
#    define HAVE_ASYNC_LDS 1
#  endif
#  if __has_builtin(__builtin_amdgcn_s_wait_asynccnt)
#    define HAVE_WAIT_ASYNC 1
#  endif
#endif

// LDS im2col offset (in halves) for im2col depth index K; K>=27 -> zero slot.
__host__ __device__ constexpr int off_of_K(int K) {
  return (K >= KK) ? ZERO_SLOT : ((K / 9) * HALO_CH + ((K % 9) / 3) * 18 + (K % 3));
}

// ---------------------------------------------------------------------------
// Kernel 1: router.  conv3x3(pad1)->global mean collapses to 9 shifted window
// sums per input channel; then 16x27 matvec, key similarity, softmax, gate.
// One thread per patch, float4 (64B-aligned) patch reads.
// ---------------------------------------------------------------------------
__global__ void router_kernel(const float* __restrict__ X,
                              const float* __restrict__ proj_w,
                              const float* __restrict__ proj_b,
                              const float* __restrict__ keys,
                              const float* __restrict__ threshold,
                              float* __restrict__ gate_out) {
  int bp = blockIdx.x * blockDim.x + threadIdx.x;
  if (bp >= BP) return;
  int b  = bp >> 8;
  int pi = bp & 255;
  int ph = pi >> 4, pw = pi & 15;

  float S[C_IN][3][3];
  #pragma unroll
  for (int c = 0; c < C_IN; ++c) {
    float T = 0.f, r0 = 0.f, r15 = 0.f, c0 = 0.f, c15 = 0.f;
    float k00 = 0.f, k015 = 0.f, k150 = 0.f, k1515 = 0.f;
    const float4* base4 = (const float4*)(X +
        ((size_t)(b * C_IN + c) * HH + ph * PATCH) * WW + pw * PATCH);
    for (int i = 0; i < PATCH; ++i) {
      float4 v4[4];
      #pragma unroll
      for (int q = 0; q < 4; ++q) v4[q] = base4[i * (WW / 4) + q];
      const float* vv = (const float*)v4;
      #pragma unroll
      for (int j = 0; j < 16; ++j) {
        float v = vv[j];
        T += v;
        if (i == 0)  r0  += v;
        if (i == 15) r15 += v;
        if (j == 0)  c0  += v;
        if (j == 15) c15 += v;
        if (j == 0)  { if (i == 0) k00  = v; if (i == 15) k150  = v; }
        if (j == 15) { if (i == 0) k015 = v; if (i == 15) k1515 = v; }
      }
    }
    // dy = ky-1: dy==-1 excludes input row 15, dy==+1 excludes input row 0.
    #pragma unroll
    for (int ky = 0; ky < 3; ++ky) {
      int dy = ky - 1;
      float er = (dy == -1) ? r15 : (dy == 1) ? r0 : 0.f;
      #pragma unroll
      for (int kx = 0; kx < 3; ++kx) {
        int dx = kx - 1;
        float ec = (dx == -1) ? c15 : (dx == 1) ? c0 : 0.f;
        float corner = 0.f;
        if (dy == -1 && dx == -1) corner = k1515;
        if (dy == -1 && dx ==  1) corner = k150;
        if (dy ==  1 && dx == -1) corner = k015;
        if (dy ==  1 && dx ==  1) corner = k00;
        S[c][ky][kx] = T - er - ec + corner;
      }
    }
  }

  float pooled[C_PROJ];
  #pragma unroll
  for (int o = 0; o < C_PROJ; ++o) {
    float acc = 0.f;
    #pragma unroll
    for (int c = 0; c < C_IN; ++c)
      #pragma unroll
      for (int ky = 0; ky < 3; ++ky)
        #pragma unroll
        for (int kx = 0; kx < 3; ++kx)
          acc += proj_w[((o * C_IN + c) * 3 + ky) * 3 + kx] * S[c][ky][kx];
    pooled[o] = acc * (1.0f / 256.0f) + proj_b[o];
  }

  float logits[EE];
  float mx = -1e30f;
  #pragma unroll
  for (int e = 0; e < EE; ++e) {
    float a = 0.f;
    #pragma unroll
    for (int o = 0; o < C_PROJ; ++o) a += pooled[o] * keys[e * C_PROJ + o];
    logits[e] = a;
    mx = fmaxf(mx, a);
  }
  float den = 0.f;
  #pragma unroll
  for (int e = 0; e < EE; ++e) { logits[e] = __expf(logits[e] - mx); den += logits[e]; }
  float thr = threshold[0];
  float g[EE], gsum = 0.f;
  #pragma unroll
  for (int e = 0; e < EE; ++e) {
    float s = logits[e] / den;
    float sig = 1.0f / (1.0f + __expf(-(s - thr) * GATE_STEEP));
    g[e] = s * sig;
    gsum += g[e];
  }
  float inv = 1.0f / (gsum + 1e-9f);
  #pragma unroll
  for (int e = 0; e < EE; ++e) gate_out[bp * EE + e] = g[e] * inv;
}

// ---------------------------------------------------------------------------
// Kernel 2: experts as per-patch im2col GEMM on the WMMA pipe.
// One block (128 thr = 4 waves) per patch; wave w owns M-tiles 4w..4w+3
// (M-tile == patch row group).  Flow:
//   (a) async-stage raw 3x16x16 patch -> LDS (gfx1250 global_load_async_to_lds)
//   (b) stage expert weights directly in WMMA B-fragment order
//   (c) build zero-haloed f16 patch; A-fragments gathered branch-free
//   (d) 16x wmma_f32_16x16x32_f16; bias+ReLU+gate accumulated in registers
//   (e) one LDS store per result element into duplicated (expert-half) slots
//   (f) chunk-parallel reduction into <=2x2 SPP 66-cell partials
// ---------------------------------------------------------------------------
__global__ void __launch_bounds__(128)
expert_kernel(const float* __restrict__ X,
              const float* __restrict__ expert_w,   // [64 * 27]
              const float* __restrict__ expert_b,   // [64]
              const float* __restrict__ gate_in,    // [BP * 8]
              float* __restrict__ part)             // [BP*8*2*2]
{
  __shared__ float    rawf[C_IN * PATCH * PATCH];                 // 3 KB
  __shared__ _Float16 rawh[RAWH_SZ];                              // 2.5 KB
  __shared__ _Float16 BfragLds[4][32][16] __attribute__((aligned(32))); // 4 KB
  __shared__ float    comb2[PP * 18];                             // 18 KB
  __shared__ float    red[4][8][4];                               // 0.5 KB
  __shared__ float    sgate[EE];
  __shared__ float    sbias[NCH];

  const int bp = blockIdx.x;
  const int b  = bp >> 8;
  const int pi = bp & 255;
  const int ph = pi >> 4, pw = pi & 15;
  const int tid = threadIdx.x;

  // ---- (a) stage raw patch rows: 384 x 8B chunks ----
  #pragma unroll
  for (int it = 0; it < 3; ++it) {
    int k = it * 128 + tid;          // chunk of 2 floats
    int e2  = k * 2;
    int c   = e2 >> 8;               // / 256
    int rem = e2 & 255;
    int row = rem >> 4, col = rem & 15;
    const float* gsrc = X + ((size_t)(b * C_IN + c) * HH + ph * PATCH + row) * WW
                          + pw * PATCH + col;
    float* ldst = &rawf[e2];
#if defined(HAVE_ASYNC_LDS)
    __builtin_amdgcn_global_load_async_to_lds_b64(
        (as1_v2i*)gsrc, (as3_v2i*)ldst, 0, 0);
#else
    ldst[0] = gsrc[0];
    ldst[1] = gsrc[1];
#endif
  }

  // ---- (b) weights -> B-fragment layout: Bfrag[nt][lane][q], K=(q<8)?h*8+q:16+h*8+q-8 ----
  for (int t = tid; t < 4 * 32 * 16; t += 128) {
    int q = t & 15, L = (t >> 4) & 31, nt = t >> 9;
    int K = (q < 8) ? ((L >> 4) * 8 + q) : (16 + (L >> 4) * 8 + (q - 8));
    int n = nt * 16 + (L & 15);
    BfragLds[nt][L][q] = (_Float16)((K < KK) ? expert_w[n * KK + K] : 0.f);
  }
  if (tid < NCH) sbias[tid] = expert_b[tid];
  if (tid < EE)  sgate[tid] = gate_in[bp * EE + tid];

#if defined(HAVE_ASYNC_LDS)
#  if defined(HAVE_WAIT_ASYNC)
  __builtin_amdgcn_s_wait_asynccnt(0);
#  else
  asm volatile("s_wait_asynccnt 0x0" ::: "memory");
#  endif
#endif
  __syncthreads();

  // ---- (c) zero-haloed f16 patch + zero tail ----
  for (int t = tid; t < RAWH_SZ; t += 128) {
    float v = 0.f;
    if (t < HALO_SZ) {
      int c = t / HALO_CH, rr = t % HALO_CH;
      int row = rr / 18, col = rr % 18;
      if (row >= 1 && row <= 16 && col >= 1 && col <= 16)
        v = rawf[(c * PATCH + row - 1) * PATCH + col - 1];
    }
    rawh[t] = (_Float16)v;
  }
  __syncthreads();

  // ---- (d) WMMA main loop ----
  const int wid  = tid >> 5;
  const int lane = tid & 31;
  const int l16  = lane & 15;
  const int h    = lane >> 4;
  const int cchan = l16 & 7;
  const int jsel  = (l16 >> 3) & 1;   // which expert-half this lane carries

  v16h  bf[4];
  float ge[4], bia[4];
  #pragma unroll
  for (int nt = 0; nt < 4; ++nt) {
    bf[nt] = *(const v16h*)&BfragLds[nt][lane][0];   // 2x ds_load_b128
    int n = nt * 16 + l16;
    ge[nt]  = sgate[n >> 3];
    bia[nt] = sbias[n];
  }

  #pragma unroll
  for (int mi = 0; mi < 4; ++mi) {
    const int mt = wid * 4 + mi;          // M-tile == patch row block; i = mt
    const int base = mt * 18 + l16;       // &rawh[0][mt][l16], halo coords
    v16h af;
    #pragma unroll
    for (int q = 0; q < 16; ++q) {
      const int K0 = (q < 8) ? q : (q + 8);     // h==0 K; h==1 is K0+8
      const int o0 = off_of_K(K0);
      const int o1 = off_of_K(K0 + 8);
      af[q] = rawh[base + (h ? o1 : o0)];       // branch-free ds_load_u16
    }

    float partial[8] = {0.f, 0.f, 0.f, 0.f, 0.f, 0.f, 0.f, 0.f};
    #pragma unroll
    for (int nt = 0; nt < 4; ++nt) {
      v8f acc = {};
      acc = __builtin_amdgcn_wmma_f32_16x16x32_f16(
          false, af, false, bf[nt], (short)0, acc, false, false);
      #pragma unroll
      for (int r = 0; r < 8; ++r) {
        float v = acc[r] + bia[nt];
        v = fmaxf(v, 0.f) * ge[nt];
        partial[r] += v;                  // sums experts e = 2*nt + jsel
      }
    }
    // ---- (e) single unconditional store per element ----
    #pragma unroll
    for (int r = 0; r < 8; ++r) {
      int p = mt * 16 + r + 8 * h;        // pixel index in patch
      comb2[p * 18 + cchan * 2 + jsel] = partial[r];
    }
  }
  __syncthreads();

  // ---- (f) SPP 66-cell partials: 128-thread chunked reduction ----
  {
    int c    = tid & 7;
    int cell = (tid >> 3) & 3;            // (ru,rv)
    int chunk = tid >> 5;                 // 4-row chunk
    int ru = cell >> 1, rv = cell & 1;
    int r0 = ph * PATCH, c0 = pw * PATCH;
    int cu0 = (r0 + 4) / 66, cv0 = (c0 + 4) / 66;
    int isplit = 16, jsplit = 16;
    for (int i = 0; i < 16; ++i) if ((r0 + i + 4) / 66 != cu0) { isplit = i; break; }
    for (int j = 0; j < 16; ++j) if ((c0 + j + 4) / 66 != cv0) { jsplit = j; break; }
    int ilo = ru ? isplit : 0, ihi = ru ? 16 : isplit;
    int ib = max(ilo, chunk * 4), ie = min(ihi, chunk * 4 + 4);
    int jb = rv ? jsplit : 0,  je = rv ? 16 : jsplit;
    float s = 0.f;
    for (int i = ib; i < ie; ++i)
      for (int j = jb; j < je; ++j) {
        int p = i * 16 + j;
        s += comb2[p * 18 + c * 2] + comb2[p * 18 + c * 2 + 1];
      }
    red[cell][c][chunk] = s;
  }
  __syncthreads();
  if (tid < 32) {
    int c = tid & 7, cell = tid >> 3;
    float s = red[cell][c][0] + red[cell][c][1] + red[cell][c][2] + red[cell][c][3];
    part[((bp * 8 + c) * 2 + (cell >> 1)) * 2 + (cell & 1)] = s;
  }
}

// ---------------------------------------------------------------------------
// Kernel 3: fold patch partials into S4[b][c][4][4] (66-cell sums of y,
// intersected with the pad-4 interior).  One thread per (b,c,u,v).
// ---------------------------------------------------------------------------
__global__ void reduce_cells_kernel(const float* __restrict__ part,
                                    float* __restrict__ S4) {
  int idx = blockIdx.x * blockDim.x + threadIdx.x;
  if (idx >= BB * 8 * 16) return;
  int v = idx & 3, u = (idx >> 2) & 3, c = (idx >> 4) & 7, b = idx >> 7;
  float s = 0.f;
  for (int ph = 0; ph < 16; ++ph) {
    int cu0 = (ph * 16 + 4) / 66, cu1 = (ph * 16 + 19) / 66;
    int ru = -1;
    if (u == cu0) ru = 0; else if (u == cu1 && cu1 != cu0) ru = 1;
    if (ru < 0) continue;
    for (int pw = 0; pw < 16; ++pw) {
      int cv0 = (pw * 16 + 4) / 66, cv1 = (pw * 16 + 19) / 66;
      int rv = -1;
      if (v == cv0) rv = 0; else if (v == cv1 && cv1 != cv0) rv = 1;
      if (rv < 0) continue;
      int bp = b * PP + ph * 16 + pw;
      s += part[((bp * 8 + c) * 2 + ru) * 2 + rv];
    }
  }
  S4[((b * 8 + c) * 4 + u) * 4 + v] = s;
}

// ---------------------------------------------------------------------------
// Kernel 4: SPP features.  spp[b][f] = final_b[o] + sum_c fw[o][c]*R/area.
// ---------------------------------------------------------------------------
__global__ void spp_kernel(const float* __restrict__ S4,
                           const float* __restrict__ final_w,
                           const float* __restrict__ final_b,
                           float* __restrict__ spp) {
  int idx = blockIdx.x * blockDim.x + threadIdx.x;
  if (idx >= BB * SPP_DIM) return;
  int b = idx / SPP_DIM, f = idx % SPP_DIM;
  int o, lvl, u = 0, v = 0;
  if (f < 8)       { lvl = 1; o = f; }
  else if (f < 40) { lvl = 2; int q = f - 8;  o = q >> 2; u = (q >> 1) & 1; v = q & 1; }
  else             { lvl = 4; int q = f - 40; o = q >> 4; u = (q >> 2) & 3; v = q & 3; }
  float inv_area = (lvl == 1) ? (1.0f / (264.f * 264.f))
                 : (lvl == 2) ? (1.0f / (132.f * 132.f))
                              : (1.0f / (66.f * 66.f));
  float acc = final_b[o];
  for (int c = 0; c < 8; ++c) {
    const float* s4 = S4 + ((b * 8 + c) * 16);
    float R = 0.f;
    if (lvl == 4)      R = s4[u * 4 + v];
    else if (lvl == 2) R = s4[(u*2)*4 + v*2] + s4[(u*2)*4 + v*2+1]
                         + s4[(u*2+1)*4 + v*2] + s4[(u*2+1)*4 + v*2+1];
    else { for (int t = 0; t < 16; ++t) R += s4[t]; }
    acc += final_w[o * 8 + c] * R * inv_area;
  }
  spp[b * SPP_DIM + f] = acc;
}

// ---------------------------------------------------------------------------
// Kernel 5: head.  out[b][n] = lin_b[n] + spp[b][:] . lin_w[:,n]
// ---------------------------------------------------------------------------
__global__ void head_kernel(const float* __restrict__ spp,
                            const float* __restrict__ lin_w,
                            const float* __restrict__ lin_b,
                            float* __restrict__ out) {
  int idx = blockIdx.x * blockDim.x + threadIdx.x;
  if (idx >= BB * NUM_CLASSES) return;
  int b = idx / NUM_CLASSES, n = idx % NUM_CLASSES;
  float s = lin_b[n];
  const float* sp = spp + b * SPP_DIM;
  for (int k = 0; k < SPP_DIM; ++k) s += sp[k] * lin_w[k * NUM_CLASSES + n];
  out[idx] = s;
}

// ---------------------------------------------------------------------------
extern "C" void kernel_launch(void* const* d_in, const int* in_sizes, int n_in,
                              void* d_out, int out_size, void* d_ws, size_t ws_size,
                              hipStream_t stream) {
  const float* X         = (const float*)d_in[0];
  const float* expert_w  = (const float*)d_in[1];
  const float* expert_b  = (const float*)d_in[2];
  const float* proj_w    = (const float*)d_in[3];
  const float* proj_b    = (const float*)d_in[4];
  const float* keys      = (const float*)d_in[5];
  const float* threshold = (const float*)d_in[6];
  const float* final_w   = (const float*)d_in[7];
  const float* final_b   = (const float*)d_in[8];
  const float* lin_w     = (const float*)d_in[9];
  const float* lin_b     = (const float*)d_in[10];
  float* out = (float*)d_out;

  float* ws   = (float*)d_ws;
  float* gate = ws;                       // BP*8      = 65536 floats
  float* part = gate + BP * 8;            // BP*8*4    = 262144 floats
  float* S4   = part + BP * 8 * 4;        // 32*8*16   = 4096 floats
  float* spp  = S4 + BB * 8 * 16;         // 32*168    = 5376 floats

  router_kernel<<<(BP + 255) / 256, 256, 0, stream>>>(
      X, proj_w, proj_b, keys, threshold, gate);

  expert_kernel<<<BP, 128, 0, stream>>>(
      X, expert_w, expert_b, gate, part);

  reduce_cells_kernel<<<(BB * 8 * 16 + 255) / 256, 256, 0, stream>>>(part, S4);

  spp_kernel<<<(BB * SPP_DIM + 255) / 256, 256, 0, stream>>>(
      S4, final_w, final_b, spp);

  head_kernel<<<(BB * NUM_CLASSES + 255) / 256, 256, 0, stream>>>(
      spp, lin_w, lin_b, out);
}
